// TwoBodySphericalHarmonicTensorEmbed_64776696758369
// MI455X (gfx1250) — matrix operand order repaired and checked by reading, not compiled, not run on hardware
//
#include <hip/hip_runtime.h>
#include <hip/hip_bf16.h>

typedef __attribute__((ext_vector_type(16))) _Float16 v16h;
typedef __attribute__((ext_vector_type(8)))  float    v8f;
typedef __attribute__((ext_vector_type(4)))  float    v4f;
typedef __attribute__((ext_vector_type(4)))  unsigned short v4us;

#define E_EDGES   250000
#define D_IN      64
#define NTOT      192            // C_OUT * 3 irreps
#define WAVES     4
#define THREADS   (WAVES * 32)
#define TILES     (E_EDGES / 16) // 15625 (exact)
#define NT_HALF   6              // 6 n-tiles (96 n) per pass, 2 passes
#define WPAD      100            // 96 + 4 pad -> conflict-free half-wave windows

__global__ __launch_bounds__(THREADS)
void TwoBodySHTensorEmbed_kernel(const float* __restrict__ edge_vec,
                                 const float* __restrict__ edge_inv,
                                 const float* __restrict__ W_lin,
                                 float* __restrict__ out) {
    // B fragments pre-swizzled into WMMA per-lane layout: [s][t][lane][j]
    __shared__ __attribute__((aligned(16))) _Float16 Bfrag[2 * 12 * 32 * 16]; // 24576 B
    __shared__ float w_st[WAVES][16][WPAD];                                   // 25600 B
    __shared__ float sh_st[WAVES][16][12];                                    //  3072 B
    __shared__ __attribute__((aligned(16))) unsigned short tab[288];          //   576 B

    const int tid = threadIdx.x;

    // ---- Stage W_lin (f32 -> f16) into B-fragment layout. W_lin is tiny (48 KB)
    // and read by every block: keep it at RT temporal hint so it stays L2-resident.
    // B 32x16 (KxN) 16-bit layout: lane = n + 16h, elem j<8 -> K = 8h + j; j>=8 -> K = 16 + 8h + (j-8).
    for (int u = tid; u < 2 * 12 * 32 * 16; u += THREADS) {
        int j  = u & 15;
        int ln = (u >> 4) & 31;
        int tt = u >> 9;            // 0..23
        int t  = tt % 12;
        int s  = tt / 12;
        int krow = 32 * s + 8 * (ln >> 4) + (j & 7) + 16 * (j >> 3);
        int n    = 16 * t + (ln & 15);
        Bfrag[u] = (_Float16)W_lin[krow * NTOT + n];
    }
    // ---- Readout LUT: for rr in [0,288): c=rr/9, k=rr%9, widx=[0,1,1,1,2,2,2,2,2][k]
    // pack byte offsets: w_off = (c*3+widx)*4 (9 bits), sh_off = k*4 (bits 9..14)
    for (int u = tid; u < 288; u += THREADS) {
        int c = u / 9, k = u - c * 9;
        int wk = (k == 0) ? 0 : ((k < 4) ? 1 : 2);
        tab[u] = (unsigned short)(((unsigned)(c * 3 + wk) * 4u) | ((unsigned)(k * 4) << 9));
    }
    __syncthreads();

    const int wv   = tid >> 5;
    const int lane = tid & 31;
    const int ti   = blockIdx.x * WAVES + wv;
    if (ti >= TILES) return;                 // wave-uniform: EXEC stays all-1s inside

    const int e0 = ti * 16;
    const int m  = lane & 15;
    const int h  = lane >> 4;

    // ---- Spherical harmonics for this tile's 16 edges (lanes 0..15) -> LDS.
    // edge_vec is a single-pass stream: non-temporal loads.
    if (h == 0) {
        const float* v = edge_vec + (size_t)(e0 + m) * 3;
        float x = __builtin_nontemporal_load(v + 0);
        float y = __builtin_nontemporal_load(v + 1);
        float z = __builtin_nontemporal_load(v + 2);
        float rn = rsqrtf(x * x + y * y + z * z);
        x *= rn; y *= rn; z *= rn;
        const float SQ3 = 1.7320508075688772f;
        const float SQ5 = 2.2360679774997896f;
        const float SQ15 = 3.8729833462074170f;
        float* sp = &sh_st[wv][m][0];
        sp[0] = 1.0f;
        sp[1] = SQ3 * x;  sp[2] = SQ3 * y;  sp[3] = SQ3 * z;
        sp[4] = SQ15 * x * z;
        sp[5] = SQ15 * x * y;
        sp[6] = SQ5 * (y * y - 0.5f * (x * x + z * z));
        sp[7] = SQ15 * y * z;
        sp[8] = 0.5f * SQ15 * (z * z - x * x);
    }

    // ---- A fragments (f32 -> f16), ISA 16-bit A 16x32 layout:
    // lane = m + 16h; elems 0..7 -> K = 32s + 8h + j ; elems 8..15 -> K = 32s + 16 + 8h + (j-8)
    // edge_inv is a single-pass 64 MB stream: non-temporal b128 loads.
    const float* arow = edge_inv + (size_t)(e0 + m) * D_IN;
    v16h afr[2];
#pragma unroll
    for (int s = 0; s < 2; ++s) {
        const float* p0 = arow + 32 * s + 8 * h;
        v4f a0 = __builtin_nontemporal_load((const v4f*)(p0));
        v4f a1 = __builtin_nontemporal_load((const v4f*)(p0 + 4));
        v4f a2 = __builtin_nontemporal_load((const v4f*)(p0 + 16));
        v4f a3 = __builtin_nontemporal_load((const v4f*)(p0 + 20));
#pragma unroll
        for (int i = 0; i < 4; ++i) {
            afr[s][i]      = (_Float16)a0[i];
            afr[s][4 + i]  = (_Float16)a1[i];
            afr[s][8 + i]  = (_Float16)a2[i];
            afr[s][12 + i] = (_Float16)a3[i];
        }
    }

    const char* wb = (const char*)&w_st[wv][0][0];
    const char* sb = (const char*)&sh_st[wv][0][0];

#pragma unroll
    for (int p = 0; p < 2; ++p) {
        // ---- GEMM: 6 n-tiles x 2 k-steps of v_wmma_f32_16x16x32_f16
        v8f acc[NT_HALF];
#pragma unroll
        for (int t = 0; t < NT_HALF; ++t) {
            v8f a = {};
#pragma unroll
            for (int s = 0; s < 2; ++s) {
                const v16h b = *(const v16h*)&Bfrag[(((s * 12 + (p * 6 + t)) * 32 + lane) * 16)];
                a = __builtin_amdgcn_wmma_f32_16x16x32_f16(
                        /*neg_a=*/false, afr[s], /*neg_b=*/false, b,
                        /*c_mod=*/(short)0, a, /*reuse_a=*/false, /*reuse_b=*/false);
            }
            acc[t] = a;
        }

        // ---- Stage accumulators: C/D layout lane holds col n=lane%16, rows r+8h
        // stride WPAD=100 floats -> half-waves hit disjoint 16-bank windows
#pragma unroll
        for (int t = 0; t < NT_HALF; ++t)
#pragma unroll
            for (int r = 0; r < 8; ++r)
                w_st[wv][r + 8 * h][16 * t + m] = acc[t][r];
        __builtin_amdgcn_wave_barrier();   // wave-synchronous LDS handoff

        // ---- Fused SH multiply + coalesced float4 stores.
        // Pass covers per-edge float range [p*288, p*288+288); 1152 float4 per tile-pass.
        // Output is a 576 MB pure write stream: non-temporal b128 stores.
        float* outp = out + (size_t)e0 * 576 + p * 288;
        int q = lane;   // u % 72  (72 float4 per edge per pass)
        int e = 0;      // u / 72
#pragma unroll 4
        for (int it = 0; it < 36; ++it) {
            v4us tv = *(const v4us*)(tab + 4 * q);   // 4 LUT entries, 8B aligned
            const char* we = wb + e * (WPAD * 4);
            const char* se = sb + e * 48;
            v4f o;
#pragma unroll
            for (int i = 0; i < 4; ++i) {
                unsigned int tvi = (unsigned int)tv[i];
                float wval = *(const float*)(we + (tvi & 0x1FFu));
                float sval = *(const float*)(se + (tvi >> 9));
                o[i] = wval * sval;
            }
            __builtin_nontemporal_store(o, (v4f*)(outp + (size_t)e * 576 + 4 * q));
            q += 32;
            int ov = (q >= 72);
            q -= 72 & (-ov);
            e += ov;
        }
        __builtin_amdgcn_wave_barrier();
    }
}

extern "C" void kernel_launch(void* const* d_in, const int* in_sizes, int n_in,
                              void* d_out, int out_size, void* d_ws, size_t ws_size,
                              hipStream_t stream) {
    (void)in_sizes; (void)n_in; (void)out_size; (void)d_ws; (void)ws_size;
    const float* edge_vec = (const float*)d_in[0];
    const float* edge_inv = (const float*)d_in[1];
    const float* W_lin    = (const float*)d_in[2];
    float* out = (float*)d_out;

    const int blocks = (TILES + WAVES - 1) / WAVES;   // 3907
    TwoBodySHTensorEmbed_kernel<<<dim3(blocks), dim3(THREADS), 0, stream>>>(
        edge_vec, edge_inv, W_lin, out);
}
